// Self_Attention_71244917506821
// MI455X (gfx1250) — matrix operand order repaired and checked by reading, not compiled, not run on hardware
//
#include <hip/hip_runtime.h>
#include <hip/hip_bf16.h>

// ---------------------------------------------------------------------------
// SAGAN self-attention for MI455X (gfx1250), wave32 WMMA.
// B=4, C=256, Cqk=16, H=W=64, N=4096.
// Attention mapping: for output position m, weights over n are
//   softmax_n( k[:,m] . q[:,n] )   and   o[:,m] = sum_n v[:,n] * w[n].
// Flash-style single pass, 64-key softmax blocks:
//   S^T tiles (n on M axis, m on N axis) via v_wmma_f32_16x16x4_f32,
//   online softmax over n, P -> bf16 B operand, O^T[c,m] accumulated with
//   v_wmma_f32_16x16x32_bf16 (fp32 accum), fused residual + gamma epilogue.
// ---------------------------------------------------------------------------

typedef float    v8f   __attribute__((ext_vector_type(8)));
typedef float    v2f   __attribute__((ext_vector_type(2)));
typedef __bf16   v16bf __attribute__((ext_vector_type(16)));
typedef __bf16   v8bf  __attribute__((ext_vector_type(8)));
typedef unsigned v8u   __attribute__((ext_vector_type(8)));

#define Bn   4
#define Cn   256
#define CQK  16
#define Nn   4096

static __device__ __forceinline__ unsigned pack_bf16x2(float a, float b) {
    unsigned short ua = __builtin_bit_cast(unsigned short, (__bf16)a);
    unsigned short ub = __builtin_bit_cast(unsigned short, (__bf16)b);
    return (unsigned)ua | ((unsigned)ub << 16);
}

// ---------------------------------------------------------------------------
// Kernel 1: q,k projections (fp32, tiny). One thread per (b, n) pixel,
// computes all 16 q and 16 k channels. Weight reads are wave-uniform -> SMEM.
// ---------------------------------------------------------------------------
__global__ __launch_bounds__(256) void sa_proj_qk(
    const float* __restrict__ x, const float* __restrict__ Wq,
    const float* __restrict__ bq, const float* __restrict__ Wk,
    const float* __restrict__ bk, float* __restrict__ qf,
    float* __restrict__ kf)
{
    int id = blockIdx.x * 256 + threadIdx.x;       // 0 .. B*N-1
    int b  = id >> 12;
    int n  = id & (Nn - 1);
    const float* xb = x + ((size_t)b * Cn) * Nn + n;

    float qa[CQK], ka[CQK];
#pragma unroll
    for (int o = 0; o < CQK; ++o) { qa[o] = bq[o]; ka[o] = bk[o]; }

#pragma unroll 4
    for (int c = 0; c < Cn; ++c) {
        float xv = xb[(size_t)c * Nn];
#pragma unroll
        for (int o = 0; o < CQK; ++o) {
            qa[o] = fmaf(Wq[o * Cn + c], xv, qa[o]);
            ka[o] = fmaf(Wk[o * Cn + c], xv, ka[o]);
        }
    }
#pragma unroll
    for (int o = 0; o < CQK; ++o) {
        qf[((size_t)b * CQK + o) * Nn + n] = qa[o];
        kf[((size_t)b * CQK + o) * Nn + n] = ka[o];
    }
}

// ---------------------------------------------------------------------------
// Kernel 2: v projection with bf16 WMMA. One wave per 16(c_out) x 16(n) tile,
// K-loop over the 256 input channels in steps of 32.
// A = Wv tile (16x32), B = x tile (32x16), D = fp32 accum (+bias), store bf16.
// ---------------------------------------------------------------------------
__global__ __launch_bounds__(128) void sa_proj_v(
    const float* __restrict__ x, const float* __restrict__ Wv,
    const float* __restrict__ bv, __bf16* __restrict__ vb)
{
    int wave = (blockIdx.x * blockDim.x + threadIdx.x) >> 5;
    int lane = threadIdx.x & 31;
    int nt = wave & 255;            // n tile (256)
    int ct = (wave >> 8) & 15;      // c_out tile (16)
    int b  = wave >> 12;            // batch (4)
    int row = lane & 15;
    int hi  = lane >> 4;
    int n = nt * 16 + row;

    v8f acc;
#pragma unroll
    for (int j = 0; j < 8; ++j) acc[j] = bv[ct * 16 + j + 8 * hi];

    for (int kc = 0; kc < Cn; kc += 32) {
        // A (16x32 bf16): lane row = c_out, lanes 0-15 hold K {0..15},
        // lanes 16-31 hold K {16..31}; contiguous fp32 -> cvt.
        const float* wrow = Wv + (size_t)(ct * 16 + row) * Cn + kc;
        v16bf A;
#pragma unroll
        for (int e = 0; e < 8; ++e) {
            A[e]     = (__bf16)wrow[hi * 16 + e];
            A[8 + e] = (__bf16)wrow[hi * 16 + 8 + e];
        }
        // B (32x16 bf16): lane col = n; VGPR v holds K pair {2v,2v+1} (+16 hi half)
        const float* xcol = x + ((size_t)b * Cn + kc) * Nn + n;
        v16bf Bm;
#pragma unroll
        for (int v = 0; v < 8; ++v) {
            int c0 = hi * 16 + 2 * v;
            Bm[2 * v]     = (__bf16)xcol[(size_t)c0 * Nn];
            Bm[2 * v + 1] = (__bf16)xcol[(size_t)(c0 + 1) * Nn];
        }
        acc = __builtin_amdgcn_wmma_f32_16x16x32_bf16(
            false, A, false, Bm, (short)0, acc, false, false);
    }
#pragma unroll
    for (int j = 0; j < 8; ++j) {
        int c = ct * 16 + j + 8 * hi;
        vb[((size_t)b * Cn + c) * Nn + n] = (__bf16)acc[j];
    }
}

// ---------------------------------------------------------------------------
// Kernel 3: flash attention + residual. One wave per (b, 16-wide m tile).
// Keys processed in blocks of 64 (4 S^T tiles) per softmax update, so the
// accumulator rescale (128 floats, emitted as v_pk_mul_f32) and the max/sum
// reductions are amortized over 64 keys instead of 32.
// ---------------------------------------------------------------------------
__global__ __launch_bounds__(128, 1) void sa_attn(
    const float* __restrict__ x, const float* __restrict__ gamma,
    const float* __restrict__ qf, const float* __restrict__ kf,
    const __bf16* __restrict__ vb, float* __restrict__ out)
{
    int wave = (blockIdx.x * blockDim.x + threadIdx.x) >> 5;
    int lane = threadIdx.x & 31;
    int mt = wave & 255;            // query tile
    int b  = wave >> 8;             // batch
    int row = lane & 15;
    int hi  = lane >> 4;
    int m = mt * 16 + row;

    const float*  qb  = qf + (size_t)b * CQK * Nn;
    const float*  kb  = kf + (size_t)b * CQK * Nn;
    const __bf16* vbb = vb + (size_t)b * Cn * Nn;

    // Loop-invariant B operand of the logit WMMA: k[:, m] tile.
    // f32 B 4x16: VGPR v holds K = v + 2*hi ; lane col = m.
    v2f bkt[4];
#pragma unroll
    for (int s = 0; s < 4; ++s) {
        bkt[s][0] = kb[(size_t)(4 * s + 0 + 2 * hi) * Nn + m];
        bkt[s][1] = kb[(size_t)(4 * s + 1 + 2 * hi) * Nn + m];
    }

    float run_max = -1e30f, run_sum = 0.f;
    v8f acc[16];
#pragma unroll
    for (int ct = 0; ct < 16; ++ct)
#pragma unroll
        for (int j = 0; j < 8; ++j) acc[ct][j] = 0.f;

    for (int nb = 0; nb < Nn; nb += 64) {
        // ---- logits: four 16x16 S^T tiles covering keys nb .. nb+63 ----
        v8f st[4];
#pragma unroll
        for (int t = 0; t < 4; ++t)
#pragma unroll
            for (int j = 0; j < 8; ++j) st[t][j] = 0.f;

#pragma unroll
        for (int s = 0; s < 4; ++s) {
#pragma unroll
            for (int t = 0; t < 4; ++t) {
                v2f a;  // A 16x4 f32: lane row = n, VGPR v -> K = v + 2*hi
                a[0] = qb[(size_t)(4 * s + 0 + 2 * hi) * Nn + nb + 16 * t + row];
                a[1] = qb[(size_t)(4 * s + 1 + 2 * hi) * Nn + nb + 16 * t + row];
                st[t] = __builtin_amdgcn_wmma_f32_16x16x4_f32(
                    false, a, false, bkt[s], (short)0, st[t], false, false);
            }
        }

        // ---- online softmax over n (per output column m = row) ----
        float tmax = -1e30f;
#pragma unroll
        for (int t = 0; t < 4; ++t)
#pragma unroll
            for (int j = 0; j < 8; ++j) tmax = fmaxf(tmax, st[t][j]);
        tmax = fmaxf(tmax, __shfl_xor(tmax, 16, 32));
        float nmax  = fmaxf(run_max, tmax);
        float alpha = __expf(run_max - nmax);
        run_max = nmax;

        float p[4][8], tsum = 0.f;
#pragma unroll
        for (int t = 0; t < 4; ++t)
#pragma unroll
            for (int j = 0; j < 8; ++j) {
                p[t][j] = __expf(st[t][j] - nmax);
                tsum += p[t][j];
            }
        tsum += __shfl_xor(tsum, 16, 32);
        run_sum = run_sum * alpha + tsum;

        // vector*scalar -> backend can emit v_pk_mul_f32 (2 f32 per op)
#pragma unroll
        for (int ct = 0; ct < 16; ++ct) acc[ct] *= alpha;

        // ---- P -> two bf16 B operands (each 32x16: K = 32-key half, N=m) ----
        // B VGPR v: lanes 0-15 K={2v,2v+1}, lanes 16-31 K={16+2v,17+2v}.
        v8u bp0, bp1;
#pragma unroll
        for (int v = 0; v < 4; ++v) {
            unsigned l0 = pack_bf16x2(p[0][2 * v], p[0][2 * v + 1]);
            unsigned h0 = pack_bf16x2(p[1][2 * v], p[1][2 * v + 1]);
            unsigned l1 = pack_bf16x2(p[2][2 * v], p[2][2 * v + 1]);
            unsigned h1 = pack_bf16x2(p[3][2 * v], p[3][2 * v + 1]);
            unsigned l0x = __shfl_xor(l0, 16, 32);
            unsigned h0x = __shfl_xor(h0, 16, 32);
            unsigned l1x = __shfl_xor(l1, 16, 32);
            unsigned h1x = __shfl_xor(h1, 16, 32);
            bp0[v]     = hi ? h0x : l0;   // K 0..7   | 16..23
            bp0[v + 4] = hi ? h0  : l0x;  // K 8..15  | 24..31
            bp1[v]     = hi ? h1x : l1;
            bp1[v + 4] = hi ? h1  : l1x;
        }
        v16bf Bp0 = __builtin_bit_cast(v16bf, bp0);
        v16bf Bp1 = __builtin_bit_cast(v16bf, bp1);

        // ---- O^T[c, m] += V[c, 64-key block] * P ----
#pragma unroll
        for (int ct = 0; ct < 16; ++ct) {
            const __bf16* vrow =
                vbb + (size_t)(ct * 16 + row) * Nn + nb + hi * 16;
            v8bf va0 = *(const v8bf*)(vrow);        // K 0..7  / 16..23
            v8bf va1 = *(const v8bf*)(vrow + 8);    // K 8..15 / 24..31
            v8bf va2 = *(const v8bf*)(vrow + 32);
            v8bf va3 = *(const v8bf*)(vrow + 40);
            v16bf Av0 = __builtin_shufflevector(va0, va1,
                0, 1, 2, 3, 4, 5, 6, 7, 8, 9, 10, 11, 12, 13, 14, 15);
            v16bf Av1 = __builtin_shufflevector(va2, va3,
                0, 1, 2, 3, 4, 5, 6, 7, 8, 9, 10, 11, 12, 13, 14, 15);
            acc[ct] = __builtin_amdgcn_wmma_f32_16x16x32_bf16(
                false, Av0, false, Bp0, (short)0, acc[ct], false, false);
            acc[ct] = __builtin_amdgcn_wmma_f32_16x16x32_bf16(
                false, Av1, false, Bp1, (short)0, acc[ct], false, false);
        }
    }

    // ---- epilogue: normalize, residual, gamma ----
    float inv = 1.0f / run_sum;
    float g   = gamma[0];
#pragma unroll
    for (int ct = 0; ct < 16; ++ct) {
#pragma unroll
        for (int j = 0; j < 8; ++j) {
            int c = ct * 16 + j + 8 * hi;
            size_t idx = ((size_t)b * Cn + c) * Nn + m;
            out[idx] = x[idx] + g * (acc[ct][j] * inv);
        }
    }
}

// ---------------------------------------------------------------------------
extern "C" void kernel_launch(void* const* d_in, const int* in_sizes, int n_in,
                              void* d_out, int out_size, void* d_ws, size_t ws_size,
                              hipStream_t stream)
{
    const float* x     = (const float*)d_in[0];
    const float* Wq    = (const float*)d_in[1];
    const float* bq    = (const float*)d_in[2];
    const float* Wk    = (const float*)d_in[3];
    const float* bk    = (const float*)d_in[4];
    const float* Wv    = (const float*)d_in[5];
    const float* bv    = (const float*)d_in[6];
    const float* gamma = (const float*)d_in[7];
    float* out = (float*)d_out;

    // workspace layout: qf (1 MB) | kf (1 MB) | v bf16 (8 MB)
    float*  qf = (float*)d_ws;
    float*  kf = qf + (size_t)Bn * CQK * Nn;
    __bf16* vb = (__bf16*)(kf + (size_t)Bn * CQK * Nn);

    // 1) q,k projections: B*N = 16384 threads
    sa_proj_qk<<<(Bn * Nn) / 256, 256, 0, stream>>>(x, Wq, bq, Wk, bk, qf, kf);

    // 2) v projection: B * (C/16) * (N/16) = 16384 waves, 4 waves/block
    sa_proj_v<<<(Bn * (Cn / 16) * (Nn / 16)) / 4, 128, 0, stream>>>(x, Wv, bv, vb);

    // 3) flash attention + residual: B * (N/16) = 1024 waves, 4 waves/block
    sa_attn<<<(Bn * (Nn / 16)) / 4, 128, 0, stream>>>(x, gamma, qf, kf, vb, out);
}
